// CrossConvBlock_10917806867059
// MI455X (gfx1250) — matrix operand order repaired and checked by reading, not compiled, not run on hardware
//
#include <hip/hip_runtime.h>

typedef _Float16 f16;
typedef __attribute__((ext_vector_type(16))) _Float16 v16h;
typedef __attribute__((ext_vector_type(8)))  float    v8f;

#define NIMG   32          // B*Sy
#define NB     4           // B
#define HW     65536       // 256*256
#define WIDTH  256

__device__ __forceinline__ int refl(int v) {
    // reflect pad of 1: -1 -> 1, 256 -> 254
    return v < 0 ? -v : (v > 255 ? 510 - v : v);
}

// -----------------------------------------------------------------------------
// Stage 0: one-shot conversion of all conv weights f32 -> f16 in the
// WMMA-B-fragment friendly layout [tap][oc][ic]. 36864 elements total,
// permanently resident in L2/WGP$ afterwards.
// -----------------------------------------------------------------------------
__global__ __launch_bounds__(256) void convert_weights_kernel(
    const float* __restrict__ w_cross,  // (32,64,3,3)
    const float* __restrict__ w_q,      // (32,32,3,3)
    const float* __restrict__ w_c,      // (32,32,3,3)
    f16* __restrict__ wcr, f16* __restrict__ wq, f16* __restrict__ wc)
{
    int idx = blockIdx.x * 256 + threadIdx.x;        // 0..36863
    if (idx < 18432) {
        int oc = idx / 576, rem = idx % 576, ic = rem / 9, t = rem % 9;
        wcr[(t * 32 + oc) * 64 + ic] = (f16)w_cross[idx];
    } else if (idx < 27648) {
        int j = idx - 18432;
        int oc = j / 288, rem = j % 288, ic = rem / 9, t = rem % 9;
        wq[(t * 32 + oc) * 32 + ic] = (f16)w_q[j];
    } else {
        int j = idx - 27648;
        int oc = j / 288, rem = j % 288, ic = rem / 9, t = rem % 9;
        wc[(t * 32 + oc) * 32 + ic] = (f16)w_c[j];
    }
}

// -----------------------------------------------------------------------------
// Stage 1: cross conv 64ic -> 32oc, 3x3 reflect, bias+ReLU -> f16 inter,
// WITH the Sy-mean fused in-register -> f16 qin.
// Block: 128 thr = 4 waves; one (b, y0, xb) tile; loops over all 8 sy.
// Query half staged ONCE (sQ); context half restaged per sy (sC).
// K-chunk 0 of each WMMA reads sQ, chunk 1 reads sC (concat order = weights).
// -----------------------------------------------------------------------------
__global__ __launch_bounds__(128) void cross_conv1_kernel(
    const float* __restrict__ query,    // (4,1,32,256,256)
    const float* __restrict__ context,  // (4,8,32,256,256) == (32,32,256,256)
    const f16*   __restrict__ wt,       // (9,32,64) f16 [t][oc][ic]
    const float* __restrict__ bias,     // (32)
    f16* __restrict__ inter,            // (32,32,256,256) f16
    f16* __restrict__ qin)              // (4,32,256,256)  f16 (mean over sy)
{
    __shared__ f16 sQ[6 * 66 * 36];     // query ch 0..31  [row][x][cpad=36]
    __shared__ f16 sC[6 * 66 * 36];     // context ch 0..31

    const int tid = threadIdx.x;
    const int bid = blockIdx.x;
    const int xb  = (bid & 3) * 64;          // block x origin
    const int y0  = ((bid >> 2) & 63) * 4;   // 4 output rows per block
    const int b   = bid >> 8;                // 0..3

    // stage query half once (x-fastest -> coalesced)
    for (int i = tid; i < 6 * 66 * 32; i += 128) {
        int seg = i / 66, x = i - seg * 66;
        int c = seg & 31, r = seg >> 5;      // r = 0..5
        int gy = refl(y0 + r - 1);
        int gx = refl(xb - 1 + x);
        sQ[(r * 66 + x) * 36 + c] = (f16)query[(b * 32 + c) * HW + gy * WIDTH + gx];
    }

    const int lane = tid & 31;
    const int wv   = tid >> 5;
    const int g    = lane >> 4;     // lane group (K-half select)
    const int l15  = lane & 15;
    const int px   = wv * 16 + l15;
    const int xo   = xb + wv * 16 + 8 * g;
    const float bz0 = bias[l15];
    const float bz1 = bias[16 + l15];

    union U { unsigned int u[8]; v16h h; };
    v8f zero = {};
    v8f qs[4][2];
    #pragma unroll
    for (int r = 0; r < 4; ++r) { qs[r][0] = zero; qs[r][1] = zero; }

    for (int sy = 0; sy < 8; ++sy) {
        const int img = b * 8 + sy;
        __syncthreads();   // previous iteration done reading sC (and sQ visible)
        for (int i = tid; i < 6 * 66 * 32; i += 128) {
            int seg = i / 66, x = i - seg * 66;
            int c = seg & 31, r = seg >> 5;
            int gy = refl(y0 + r - 1);
            int gx = refl(xb - 1 + x);
            sC[(r * 66 + x) * 36 + c] =
                (f16)context[(img * 32 + c) * HW + gy * WIDTH + gx];
        }
        __syncthreads();

        v8f acc[4][2];
        #pragma unroll
        for (int r = 0; r < 4; ++r) { acc[r][0] = zero; acc[r][1] = zero; }

        for (int t = 0; t < 9; ++t) {
            const int ty = t / 3, tx = t % 3;
            const int lx = px + tx;
            // 4 B fragments: (kchunk 0/1) x (octile 0/1); K = 16g + 2i
            U B00, B01, B10, B11;
            #pragma unroll
            for (int i = 0; i < 8; ++i) {
                int K0 = 16 * g + 2 * i;
                const f16* w0 = &wt[(t * 32 +      l15) * 64 + K0];
                const f16* w1 = &wt[(t * 32 + 16 + l15) * 64 + K0];
                B00.u[i] = *(const unsigned int*)(w0);
                B01.u[i] = *(const unsigned int*)(w1);
                B10.u[i] = *(const unsigned int*)(w0 + 32);
                B11.u[i] = *(const unsigned int*)(w1 + 32);
            }
            #pragma unroll
            for (int row = 0; row < 4; ++row) {
                const int base = ((row + ty) * 66 + lx) * 36;
                U A0, A1;
                #pragma unroll
                for (int i = 0; i < 8; ++i) {
                    int K0 = 16 * g + 2 * i;
                    A0.u[i] = *(const unsigned int*)&sQ[base + K0];
                    A1.u[i] = *(const unsigned int*)&sC[base + K0];
                }
                acc[row][0] = __builtin_amdgcn_wmma_f32_16x16x32_f16(
                    false, A0.h, false, B00.h, (short)0, acc[row][0], false, false);
                acc[row][1] = __builtin_amdgcn_wmma_f32_16x16x32_f16(
                    false, A0.h, false, B01.h, (short)0, acc[row][1], false, false);
                acc[row][0] = __builtin_amdgcn_wmma_f32_16x16x32_f16(
                    false, A1.h, false, B10.h, (short)0, acc[row][0], false, false);
                acc[row][1] = __builtin_amdgcn_wmma_f32_16x16x32_f16(
                    false, A1.h, false, B11.h, (short)0, acc[row][1], false, false);
            }
        }

        // bias + ReLU, store inter, accumulate running sum for the mean
        #pragma unroll
        for (int row = 0; row < 4; ++row) {
            const int y = y0 + row;
            union { f16 h[8]; uint4 v4; } S;
            #pragma unroll
            for (int i = 0; i < 8; ++i) {
                float v = acc[row][0][i] + bz0;
                v = v > 0.f ? v : 0.f;
                qs[row][0][i] += v;
                S.h[i] = (f16)v;
            }
            *(uint4*)&inter[(size_t)(img * 32 + l15) * HW + y * WIDTH + xo] = S.v4;
            #pragma unroll
            for (int i = 0; i < 8; ++i) {
                float v = acc[row][1][i] + bz1;
                v = v > 0.f ? v : 0.f;
                qs[row][1][i] += v;
                S.h[i] = (f16)v;
            }
            *(uint4*)&inter[(size_t)(img * 32 + 16 + l15) * HW + y * WIDTH + xo] = S.v4;
        }
    }

    // fused mean over sy -> qin
    #pragma unroll
    for (int row = 0; row < 4; ++row) {
        const int y = y0 + row;
        union { f16 h[8]; uint4 v4; } S;
        #pragma unroll
        for (int i = 0; i < 8; ++i) S.h[i] = (f16)(qs[row][0][i] * 0.125f);
        *(uint4*)&qin[(size_t)(b * 32 + l15) * HW + y * WIDTH + xo] = S.v4;
        #pragma unroll
        for (int i = 0; i < 8; ++i) S.h[i] = (f16)(qs[row][1][i] * 0.125f);
        *(uint4*)&qin[(size_t)(b * 32 + 16 + l15) * HW + y * WIDTH + xo] = S.v4;
    }
}

// -----------------------------------------------------------------------------
// Stage 2: 32->32 ch 3x3 reflect conv + bias + ReLU, f16 in, f32 out.
// ROWS=4 per block (6 staged rows); weights from preconverted global f16.
// -----------------------------------------------------------------------------
__global__ __launch_bounds__(128) void conv32_kernel(
    const f16*   __restrict__ in,    // (nimg,32,256,256) f16
    const f16*   __restrict__ wt,    // (9,32,32) f16 [t][oc][ic]
    const float* __restrict__ bias,  // (32)
    float* __restrict__ out)         // (nimg,32,256,256) f32
{
    __shared__ f16 sIn[6 * 66 * 36];  // [row 0..5][x][cpad=36]

    const int tid = threadIdx.x;
    const int bid = blockIdx.x;
    const int xb  = (bid & 3) * 64;
    const int y0  = ((bid >> 2) & 63) * 4;
    const int img = bid >> 8;

    for (int i = tid; i < 6 * 66 * 32; i += 128) {
        int seg = i / 66, x = i - seg * 66;
        int c = seg & 31, r = seg >> 5;      // r = 0..5
        int gy = refl(y0 + r - 1);
        int gx = refl(xb - 1 + x);
        sIn[(r * 66 + x) * 36 + c] = in[(size_t)(img * 32 + c) * HW + gy * WIDTH + gx];
    }
    __syncthreads();

    const int lane = tid & 31;
    const int wv   = tid >> 5;
    const int g    = lane >> 4;
    const int l15  = lane & 15;
    const int px   = wv * 16 + l15;

    union U { unsigned int u[8]; v16h h; };
    v8f zero = {};
    v8f acc[4][2];
    #pragma unroll
    for (int r = 0; r < 4; ++r) { acc[r][0] = zero; acc[r][1] = zero; }

    for (int t = 0; t < 9; ++t) {
        const int ty = t / 3, tx = t % 3;
        const int lx = px + tx;
        U B0, B1;
        #pragma unroll
        for (int i = 0; i < 8; ++i) {
            int K0 = 16 * g + 2 * i;
            B0.u[i] = *(const unsigned int*)&wt[(t * 32 +      l15) * 32 + K0];
            B1.u[i] = *(const unsigned int*)&wt[(t * 32 + 16 + l15) * 32 + K0];
        }
        #pragma unroll
        for (int row = 0; row < 4; ++row) {
            U A;
            #pragma unroll
            for (int i = 0; i < 8; ++i) {
                int K0 = 16 * g + 2 * i;
                A.u[i] = *(const unsigned int*)&sIn[((row + ty) * 66 + lx) * 36 + K0];
            }
            acc[row][0] = __builtin_amdgcn_wmma_f32_16x16x32_f16(
                false, A.h, false, B0.h, (short)0, acc[row][0], false, false);
            acc[row][1] = __builtin_amdgcn_wmma_f32_16x16x32_f16(
                false, A.h, false, B1.h, (short)0, acc[row][1], false, false);
        }
    }

    const int xo = xb + wv * 16 + 8 * g;
    const float bz0 = bias[l15];
    const float bz1 = bias[16 + l15];
    #pragma unroll
    for (int row = 0; row < 4; ++row) {
        const int y = y0 + row;
        float* o0 = &out[(size_t)(img * 32 + l15) * HW + y * WIDTH + xo];
        float* o1 = &out[(size_t)(img * 32 + 16 + l15) * HW + y * WIDTH + xo];
        float4 s0, s1;
        s0.x = fmaxf(acc[row][0][0] + bz0, 0.f); s0.y = fmaxf(acc[row][0][1] + bz0, 0.f);
        s0.z = fmaxf(acc[row][0][2] + bz0, 0.f); s0.w = fmaxf(acc[row][0][3] + bz0, 0.f);
        s1.x = fmaxf(acc[row][0][4] + bz0, 0.f); s1.y = fmaxf(acc[row][0][5] + bz0, 0.f);
        s1.z = fmaxf(acc[row][0][6] + bz0, 0.f); s1.w = fmaxf(acc[row][0][7] + bz0, 0.f);
        *(float4*)(o0)     = s0;
        *(float4*)(o0 + 4) = s1;
        s0.x = fmaxf(acc[row][1][0] + bz1, 0.f); s0.y = fmaxf(acc[row][1][1] + bz1, 0.f);
        s0.z = fmaxf(acc[row][1][2] + bz1, 0.f); s0.w = fmaxf(acc[row][1][3] + bz1, 0.f);
        s1.x = fmaxf(acc[row][1][4] + bz1, 0.f); s1.y = fmaxf(acc[row][1][5] + bz1, 0.f);
        s1.z = fmaxf(acc[row][1][6] + bz1, 0.f); s1.w = fmaxf(acc[row][1][7] + bz1, 0.f);
        *(float4*)(o1)     = s0;
        *(float4*)(o1 + 4) = s1;
    }
}

// -----------------------------------------------------------------------------
extern "C" void kernel_launch(void* const* d_in, const int* in_sizes, int n_in,
                              void* d_out, int out_size, void* d_ws, size_t ws_size,
                              hipStream_t stream) {
    const float* query   = (const float*)d_in[0];
    const float* context = (const float*)d_in[1];
    const float* w_cross = (const float*)d_in[2];
    const float* b_cross = (const float*)d_in[3];
    const float* w_q     = (const float*)d_in[4];
    const float* b_q     = (const float*)d_in[5];
    const float* w_c     = (const float*)d_in[6];
    const float* b_c     = (const float*)d_in[7];

    float* out = (float*)d_out;
    const size_t interBytes = (size_t)NIMG * 32 * HW * 2;   // 128 MiB
    const size_t qinBytes   = (size_t)NB   * 32 * HW * 2;   //  16 MiB
    f16* inter = (f16*)d_ws;
    f16* qin   = (f16*)((char*)d_ws + interBytes);
    f16* wcr   = (f16*)((char*)d_ws + interBytes + qinBytes);   // 9*32*64
    f16* wq    = wcr + 9 * 32 * 64;                             // 9*32*32
    f16* wc    = wq  + 9 * 32 * 32;                             // 9*32*32

    // Stage 0: weight conversion (36864 elements)
    convert_weights_kernel<<<144, 256, 0, stream>>>(w_cross, w_q, w_c, wcr, wq, wc);

    // Stage 1: cross conv over all sy with fused support-mean
    cross_conv1_kernel<<<NB * 64 * 4, 128, 0, stream>>>(
        query, context, wcr, b_cross, inter, qin);

    // Stage 2a: query branch conv -> d_out[0 : 4*32*HW]
    conv32_kernel<<<NB * 64 * 4, 128, 0, stream>>>(qin, wq, b_q, out);

    // Stage 2b: context branch conv -> d_out[4*32*HW : ]
    conv32_kernel<<<NIMG * 64 * 4, 128, 0, stream>>>(
        inter, wc, b_c, out + (size_t)NB * 32 * HW);
}
